// OuterProductMean_75222057222539
// MI455X (gfx1250) — matrix operand order repaired and checked by reading, not compiled, not run on hardware
//
#include <hip/hip_runtime.h>

typedef __attribute__((ext_vector_type(16))) _Float16 v16h;
typedef __attribute__((ext_vector_type(8)))  float    v8f;

#define S_DIM 64
#define N_DIM 384
#define CIN   64
#define CH    32
#define COUT  128
#define KH    (CH * CH)   // 1024

#define JT_PER_BLOCK 8
// dynamic LDS layout (bytes):
//   [0, 262144)          WoS : 128x1024 f16 (k-permuted: k' = d*32+c)
//   BS0 [262144, +16K)   Bs ping     \   Zs (32KB) = [Bs_cur, Zx] or [Zx, Bs_cur]
//   ZX  [278528, +16K)   Z extra     /   -> always contiguous 32KB
//   BS1 [294912, +16K)   Bs pong
//   INVD [311296, +64)
#define LDS_WO_BYTES 262144
#define BS0_OFF      262144
#define ZX_OFF       278528
#define BS1_OFF      294912
#define INVD_OFF     311296
#define LDS_TOTAL    311360

union FragH { v16h v; uint4 u[2]; };

// ---------------------------------------------------------------------------
// Kernel 0: convert Wo to f16 with k-permutation: WoH[o][d*32+c] = Wo[o][c*32+d].
// (Zs is written d-major so its transpose vectorizes to b128 LDS stores; the
// contraction is invariant to a shared permutation of k.)
// ---------------------------------------------------------------------------
__global__ __launch_bounds__(256) void opm_cvt_wo(const float* __restrict__ Wo,
                                                  _Float16* __restrict__ WoH) {
    int i = blockIdx.x * blockDim.x + threadIdx.x;   // output index
    if (i < COUT * KH) {
        const int rem = i & 1023;
        const int d = rem >> 5, c = rem & 31;
        WoH[i] = (_Float16)Wo[(i & ~1023) + c * 32 + d];
    }
}

// ---------------------------------------------------------------------------
// Kernel 1: fused LayerNorm + dual projection + mask.
// One wave per (s,n) row. Outputs aT/bT laid out as (n*32+c, s) row-major f16
// so both GEMM1 operands have K (=s) contiguous per row.
// ---------------------------------------------------------------------------
__global__ __launch_bounds__(256) void opm_prep(const float* __restrict__ m,
                                                const int*   __restrict__ mask,
                                                const float* __restrict__ gamma,
                                                const float* __restrict__ beta,
                                                const float* __restrict__ Wa,
                                                const float* __restrict__ Wb,
                                                _Float16* __restrict__ aT,
                                                _Float16* __restrict__ bT) {
    __shared__ float mnS[8][CIN];
    const int wv   = threadIdx.x >> 5;
    const int lane = threadIdx.x & 31;
    const int row  = blockIdx.x * 8 + wv;          // row in [0, S*N)

    const float2 x = ((const float2*)(m + (size_t)row * CIN))[lane];
    float s1 = x.x + x.y;
    float s2 = x.x * x.x + x.y * x.y;
#pragma unroll
    for (int off = 16; off > 0; off >>= 1) {
        s1 += __shfl_xor(s1, off, 32);
        s2 += __shfl_xor(s2, off, 32);
    }
    const float mean = s1 * (1.0f / CIN);
    const float var  = s2 * (1.0f / CIN) - mean * mean;
    const float rstd = rsqrtf(var + 1e-5f);
    mnS[wv][2 * lane]     = (x.x - mean) * rstd * gamma[2 * lane]     + beta[2 * lane];
    mnS[wv][2 * lane + 1] = (x.y - mean) * rstd * gamma[2 * lane + 1] + beta[2 * lane + 1];
    __syncthreads();

    const float mf = (mask[row] != 0) ? 1.0f : 0.0f;
    const int c = lane;                             // hidden channel, 0..31
    float accA = 0.0f, accB = 0.0f;
#pragma unroll
    for (int k = 0; k < CIN; ++k) {
        const float v = mnS[wv][k];
        accA = fmaf(v, Wa[c * CIN + k], accA);
        accB = fmaf(v, Wb[c * CIN + k], accB);
    }
    const int s = row / N_DIM;
    const int n = row - s * N_DIM;
    const size_t orow = ((size_t)n * CH + c) * S_DIM + s;
    aT[orow] = (_Float16)(accA * mf);
    bT[orow] = (_Float16)(accB * mf);
}

// async prefetch of one 16KB Bs slice into LDS (4 x b128 per thread)
static __device__ __forceinline__ void prefetch_bs_async(const _Float16* bT, int j0,
                                                         uint32_t dstOff, int tid) {
    const char* src = (const char*)bT + (size_t)j0 * 128 * 128;   // contiguous 16KB
#pragma unroll
    for (int it = 0; it < 4; ++it) {
        const uint32_t t = (uint32_t)tid + it * 256u;
        const uint32_t lds_addr = dstOff + t * 16u;
        const void* g = src + (size_t)t * 16u;
        asm volatile("global_load_async_to_lds_b128 %0, %1, off"
                     :: "v"(lds_addr), "v"(g) : "memory");
    }
}

static __device__ __forceinline__ void load_z(FragH& f, const uint4* zs,
                                              int lr, int half, int ks) {
    const int aidx = lr * 128 + ks * 4 + half;        // Zs row pitch 128 uint4
    f.u[0] = zs[aidx];
    f.u[1] = zs[aidx + 2];
}
static __device__ __forceinline__ void load_w(FragH& f, const uint4* woS,
                                              int o, int half, int ks) {
    const int widx = o * 128 + ks * 4 + half;         // Wo row pitch 128 uint4
    f.u[0] = woS[widx];
    f.u[1] = woS[widx + 2];
}

// ---------------------------------------------------------------------------
// Kernel 2: Wo LDS-resident (256 KB of 320 KB WGP LDS); per-wave GEMM1
// A-fragments live in VGPRs for the whole block; Bs double-buffered with
// GLOBAL_LOAD_ASYNC_TO_LDS_B128 (ASYNCcnt); GEMM2 runs two accumulator
// chains with depth-2 fragment pipelining so ds_load latency hides under WMMA.
// ---------------------------------------------------------------------------
__global__ __launch_bounds__(256) void opm_main(const _Float16* __restrict__ aT,
                                                const _Float16* __restrict__ bT,
                                                const _Float16* __restrict__ WoH,
                                                const int*      __restrict__ mask,
                                                const float*    __restrict__ bo,
                                                float* __restrict__ out) {
    extern __shared__ char smem_dyn[];
    uint4* woS   = (uint4*)smem_dyn;                       // 128x1024 f16
    float* invdS = (float*)(smem_dyn + INVD_OFF);          // 16 floats

    const int tid  = threadIdx.x;
    const int wv   = tid >> 5;
    const int lane = tid & 31;
    const int half = lane >> 4;
    const int lr   = lane & 15;
    const int i0 = blockIdx.x;                             // 0..95
    const uint32_t ldsBase = __builtin_amdgcn_groupstaticsize();  // dyn LDS base

    // ---- kick off async prefetch of the first Bs slice into BS0 ----
    prefetch_bs_async(bT, blockIdx.y * JT_PER_BLOCK, ldsBase + BS0_OFF, tid);

    // ---- stage Wo (f16, k-permuted (o,k')) into LDS: 16384 uint4 ----
    {
        const uint4* gW = (const uint4*)WoH;
#pragma unroll
        for (int it = 0; it < 64; ++it) {
            const int idx = tid + it * 256;
            woS[idx] = gW[idx];
        }
    }

    // ---- per-wave GEMM1 A-fragments from global, invariant over j-tiles ----
    FragH afragK[2];
#pragma unroll
    for (int ks = 0; ks < 2; ++ks) {
        const char* rowp = (const char*)aT +
            (size_t)(i0 * 128 + wv * 16 + lr) * 128 + ks * 64 + half * 16;
        afragK[ks].u[0] = *(const uint4*)rowp;
        afragK[ks].u[1] = *(const uint4*)(rowp + 32);
    }

    const int o = wv * 16 + lr;            // this wave's output channel
    const float bov = bo[o];

    for (int jt = 0; jt < JT_PER_BLOCK; ++jt) {
        const int j0  = blockIdx.y * JT_PER_BLOCK + jt;    // 0..95
        const int cur = jt & 1;
        const uint32_t bsOff = cur ? BS1_OFF : BS0_OFF;
        const uint32_t zsOff = cur ? ZX_OFF  : BS0_OFF;    // contiguous 32KB

        // Bs[cur] complete + all LDS reads of previous iteration finished
        asm volatile("s_wait_asynccnt 0x0" ::: "memory");
        __syncthreads();

        // ---- pairwise mask denominator for the 16 (i,j) pairs ----
        if (tid < 16) {
            const int i = i0 * 4 + (tid >> 2);
            const int j = j0 * 4 + (tid & 3);
            int cnt = 0;
#pragma unroll
            for (int s = 0; s < S_DIM; ++s)
                cnt += (mask[s * N_DIM + i] != 0) & (mask[s * N_DIM + j] != 0);
            invdS[tid] = 1.0f / fmaxf((float)cnt, 1.0f);
        }

        // ---- GEMM1: z(128x128) = A(regs) * Bs(128x64)^T, K = s = 64 ----
        const uint4* bs = (const uint4*)(smem_dyn + bsOff);
        v8f acc[8] = {};
#pragma unroll
        for (int ks = 0; ks < 2; ++ks) {
#pragma unroll
            for (int t = 0; t < 8; ++t) {
                FragH bfrag;   // B frag: half h -> K h*8..h*8+7, 16+h*8..16+h*8+7
                const int idx = (t * 16 + lr) * 8 + ks * 4 + half;
                bfrag.u[0] = bs[idx];
                bfrag.u[1] = bs[idx + 2];
                acc[t] = __builtin_amdgcn_wmma_f32_16x16x32_f16(
                    false, afragK[ks].v, false, bfrag.v, (short)0, acc[t],
                    false, false);
            }
        }

        // ---- prefetch next Bs slice into the other buffer ----
        if (jt + 1 < JT_PER_BLOCK)
            prefetch_bs_async(bT, j0 + 1, ldsBase + (cur ? BS0_OFF : BS1_OFF), tid);

        __syncthreads();   // GEMM1 done reading Bs[cur]; invd visible

        // ---- scale by 1/denom, cvt f16, transpose into Zs[pair][d*32+c] ----
        // Per (wave,half) the 8 D-rows are 8 consecutive c values, so each
        // tile packs into one aligned b128 LDS store.
        {
            _Float16* Zs = (_Float16*)(smem_dyn + zsOff);   // 16 x 1024 f16
            const int mbase = wv * 16 + 8 * half;
            const int i_l   = mbase >> 5;
            const int cbase = mbase & 31;
#pragma unroll
            for (int t = 0; t < 8; ++t) {
                const int ncol = t * 16 + lr;               // j_l*32 + d
                const int pair = (i_l << 2) + (ncol >> 5);
                const float inv = invdS[pair];
                union { _Float16 h[8]; uint4 u; } pk;
#pragma unroll
                for (int r = 0; r < 8; ++r)
                    pk.h[r] = (_Float16)(acc[t][r] * inv);
                *(uint4*)(Zs + pair * KH + (ncol & 31) * 32 + cbase) = pk.u;
            }
        }
        __syncthreads();

        // ---- GEMM2: out(16x128) = Zs(16x1024) * Wo^T; two acc chains,
        //      fragments pipelined 2 k-steps ahead ----
        const uint4* zs = (const uint4*)(smem_dyn + zsOff);
        v8f dacc0 = {}, dacc1 = {};
        FragH za0, wb0, za1, wb1;
        load_z(za0, zs, lr, half, 0);  load_w(wb0, woS, o, half, 0);
        load_z(za1, zs, lr, half, 1);  load_w(wb1, woS, o, half, 1);
#pragma unroll
        for (int ks = 0; ks < 32; ks += 2) {
            FragH zaN0, wbN0, zaN1, wbN1;
            if (ks + 2 < 32) {
                load_z(zaN0, zs, lr, half, ks + 2);  load_w(wbN0, woS, o, half, ks + 2);
                load_z(zaN1, zs, lr, half, ks + 3);  load_w(wbN1, woS, o, half, ks + 3);
            }
            dacc0 = __builtin_amdgcn_wmma_f32_16x16x32_f16(
                false, za0.v, false, wb0.v, (short)0, dacc0, false, false);
            dacc1 = __builtin_amdgcn_wmma_f32_16x16x32_f16(
                false, za1.v, false, wb1.v, (short)0, dacc1, false, false);
            if (ks + 2 < 32) {
                za0 = zaN0; wb0 = wbN0; za1 = zaN1; wb1 = wbN1;
            }
        }
        const v8f dacc = dacc0 + dacc1;

        // ---- epilogue: + bo, store f32 out[(i*384+j)*128 + o] ----
#pragma unroll
        for (int r = 0; r < 8; ++r) {
            const int pair = r + 8 * half;          // D row = pair index
            const int i_l = pair >> 2, j_l = pair & 3;
            const size_t oidx =
                ((size_t)(i0 * 4 + i_l) * N_DIM + (j0 * 4 + j_l)) * COUT + o;
            out[oidx] = dacc[r] + bov;
        }
    }
}

// ---------------------------------------------------------------------------
extern "C" void kernel_launch(void* const* d_in, const int* in_sizes, int n_in,
                              void* d_out, int out_size, void* d_ws, size_t ws_size,
                              hipStream_t stream) {
    (void)in_sizes; (void)n_in; (void)out_size; (void)ws_size;
    const float* m     = (const float*)d_in[0];
    const int*   mask  = (const int*)d_in[1];
    const float* gamma = (const float*)d_in[2];
    const float* beta  = (const float*)d_in[3];
    const float* Wa    = (const float*)d_in[4];
    const float* Wb    = (const float*)d_in[5];
    const float* Wo    = (const float*)d_in[6];
    const float* bo    = (const float*)d_in[7];
    float* out = (float*)d_out;

    // workspace: aT (1.5 MB) | bT (1.5 MB) | WoH (256 KB)  -- all 16B aligned
    _Float16* aT  = (_Float16*)d_ws;
    _Float16* bT  = aT + (size_t)N_DIM * CH * S_DIM;
    _Float16* WoH = bT + (size_t)N_DIM * CH * S_DIM;

    opm_cvt_wo<<<(COUT * KH + 255) / 256, 256, 0, stream>>>(Wo, WoH);
    opm_prep<<<(S_DIM * N_DIM) / 8, 256, 0, stream>>>(m, mask, gamma, beta, Wa, Wb, aT, bT);
    dim3 grid(N_DIM / 4, N_DIM / (4 * JT_PER_BLOCK));
    opm_main<<<grid, 256, LDS_TOTAL, stream>>>(aT, bT, WoH, mask, bo, out);
}
